// MultiHeadAttention_11647951307228
// MI455X (gfx1250) — matrix operand order repaired and checked by reading, not compiled, not run on hardware
//
#include <hip/hip_runtime.h>

// Problem constants (match reference)
constexpr int BATCH = 2;
constexpr int L = 2048;
constexpr int D = 1024;
constexpr int H = 16;
constexpr int HD = 64;       // head dim

typedef __attribute__((ext_vector_type(16))) __bf16 v16bf;
typedef __attribute__((ext_vector_type(8)))  float  v8f;

union BfVec {
    v16bf v;
    uint4 q[2];
};

// round-to-nearest-even f32 -> bf16 (as raw u16)
__device__ __host__ inline unsigned short f2bf(float f) {
    unsigned u = __builtin_bit_cast(unsigned, f);
    unsigned r = u + 0x7FFFu + ((u >> 16) & 1u);
    return (unsigned short)(r >> 16);
}

// ---------------------------------------------------------------------------
// WMMA operand loaders, derived from CDNA5 ISA VGPR layouts (CDNA5 ISA):
//  A (16x32 bf16): lane l holds row M=l&15; K(e) = e + (e>=8 ? 8 : 0) + (l>=16 ? 8 : 0)
//  B (32x16 bf16): lane l holds col N=l&15; K(e) = e + (e>=8 ? 8 : 0) + (l>=16 ? 8 : 0)
//    -> needs "n-major" memory (contraction dim contiguous): mem[n][k], leading dim ld
//  C/D (16x16 f32): lane l holds col N=l&15, VGPR i holds row M = i + (l>=16 ? 8 : 0)
// ---------------------------------------------------------------------------
__device__ inline v16bf load_a_tile(const unsigned short* src, int ld, int row0, int col0, int lane) {
    int ln = lane & 15, half = lane >> 4;
    const unsigned short* p = src + (size_t)(row0 + ln) * ld + col0 + half * 8;
    BfVec r;
    r.q[0] = *(const uint4*)(p);        // K = half*8 + 0..7
    r.q[1] = *(const uint4*)(p + 16);   // K = half*8 + 16..23
    return r.v;
}

__device__ inline v16bf load_b_tile(const unsigned short* src, int ld, int n0, int k0, int lane) {
    int ln = lane & 15, half = lane >> 4;
    const unsigned short* p = src + (size_t)(n0 + ln) * ld + k0 + half * 8;
    BfVec r;
    r.q[0] = *(const uint4*)(p);        // K = half*8 + 0..7
    r.q[1] = *(const uint4*)(p + 16);   // K = half*8 + 16..23
    return r.v;
}

__device__ inline v8f wmma_bf16(v16bf a, v16bf b, v8f c) {
    return __builtin_amdgcn_wmma_f32_16x16x32_bf16(
        /*neg_a=*/false, a, /*neg_b=*/false, b,
        /*c_mod=*/(short)0, c, /*reuse_a=*/false, /*reuse_b=*/false);
}

// CDNA5 async copy: per-lane 16B global -> LDS, tracked by ASYNCcnt.
__device__ inline void async_cp16(unsigned lds_off, const void* gaddr) {
    asm volatile("global_load_async_to_lds_b128 %0, %1, off"
                 :: "v"(lds_off), "v"(gaddr)
                 : "memory");
}

// ---------------------------------------------------------------------------
// Stage 0: dtype conversion / weight transposition
// ---------------------------------------------------------------------------
__global__ void cvt_f32_to_bf16(const float* __restrict__ src, unsigned short* __restrict__ dst, int n) {
    int i = blockIdx.x * blockDim.x + threadIdx.x;
    if (i < n) dst[i] = f2bf(src[i]);
}

// src (G,R,C) f32 -> dst (G,C,R) bf16   (puts contraction dim R contiguous)
__global__ void transpose_to_bf16(const float* __restrict__ src, unsigned short* __restrict__ dst,
                                  int G, int R, int C) {
    int i = blockIdx.x * blockDim.x + threadIdx.x;
    int total = G * R * C;
    if (i < total) {
        int g = i / (R * C);
        int rem = i % (R * C);
        int r = rem / C;
        int c = rem % C;
        dst[(size_t)g * R * C + (size_t)c * R + r] = f2bf(src[i]);
    }
}

// ---------------------------------------------------------------------------
// Stage 1: fused QKV projection.  One wave -> 16 rows x 64 cols of Q, K, V
// for one head.  Q,K stored (B,H,L,HD) bf16; V stored transposed (B,H,HD,L).
// ---------------------------------------------------------------------------
__global__ __launch_bounds__(32) void qkv_kernel(
    const unsigned short* __restrict__ xb,                         // (B*L, D) bf16
    const unsigned short* __restrict__ wqT,                        // (H, HD, D) bf16
    const unsigned short* __restrict__ wkT,
    const unsigned short* __restrict__ wvT,
    const float* __restrict__ bq, const float* __restrict__ bk, const float* __restrict__ bv,
    unsigned short* __restrict__ Qb, unsigned short* __restrict__ Kb,
    unsigned short* __restrict__ Vt) {
    int lane = threadIdx.x;
    int ln = lane & 15, half = lane >> 4;
    int row0 = blockIdx.x * 16;        // row tile in (B*L)
    int h = blockIdx.y;

    const unsigned short* wq = wqT + (size_t)h * HD * D;
    const unsigned short* wk = wkT + (size_t)h * HD * D;
    const unsigned short* wv = wvT + (size_t)h * HD * D;

    v8f aq[4] = {}, ak[4] = {}, av[4] = {};
    for (int k0 = 0; k0 < D; k0 += 32) {
        v16bf a = load_a_tile(xb, D, row0, k0, lane);
#pragma unroll
        for (int j = 0; j < 4; j++) {
            aq[j] = wmma_bf16(a, load_b_tile(wq, D, j * 16, k0, lane), aq[j]);
            ak[j] = wmma_bf16(a, load_b_tile(wk, D, j * 16, k0, lane), ak[j]);
            av[j] = wmma_bf16(a, load_b_tile(wv, D, j * 16, k0, lane), av[j]);
        }
    }
#pragma unroll
    for (int j = 0; j < 4; j++) {
#pragma unroll
        for (int i = 0; i < 8; i++) {
            int row = row0 + i + half * 8;   // global row in (B*L)
            int bb = row / L;
            int q = row % L;
            int e = j * 16 + ln;
            size_t qkoff = (((size_t)(bb * H + h)) * L + q) * HD + e;
            Qb[qkoff] = f2bf(aq[j][i] + bq[h * HD + e]);
            Kb[qkoff] = f2bf(ak[j][i] + bk[h * HD + e]);
            Vt[(((size_t)(bb * H + h)) * HD + e) * L + q] = f2bf(av[j][i] + bv[h * HD + e]);
        }
    }
}

// ---------------------------------------------------------------------------
// Stage 2: flash attention, 4 waves / workgroup (64 q rows of one (b,h)).
// K/V key-blocks (32 keys) are staged into LDS with CDNA5 async copies
// (GLOBAL_LOAD_ASYNC_TO_LDS_B128, ASYNCcnt double-buffered); all 4 waves
// consume the shared staged block.  Writes concat (B, L, D) bf16.
// ---------------------------------------------------------------------------
__global__ __launch_bounds__(128) void attn_kernel(
    const unsigned short* __restrict__ Qb,   // (B,H,L,HD)
    const unsigned short* __restrict__ Kb,   // (B,H,L,HD)
    const unsigned short* __restrict__ Vt,   // (B,H,HD,L)
    const int* __restrict__ mask,            // (B,L)
    unsigned short* __restrict__ concat) {   // (B,L,D)
    __shared__ unsigned short sK[2][32 * HD];   // [buf][key 32 x HD]   ld = HD
    __shared__ unsigned short sV[2][HD * 32];   // [buf][HD x key 32]   ld = 32
    __shared__ unsigned short sP[4][16 * 32];   // per-wave P bounce (C->A layout)

    int tid = threadIdx.x;
    int lane = tid & 31;
    int wave = tid >> 5;
    int ln = lane & 15, half = lane >> 4;
    int q0 = blockIdx.x * 64 + wave * 16;
    int h = blockIdx.y;
    int b = blockIdx.z;

    const unsigned short* Qh = Qb + ((size_t)(b * H + h)) * L * HD;
    const unsigned short* Kh = Kb + ((size_t)(b * H + h)) * L * HD;
    const unsigned short* Vh = Vt + ((size_t)(b * H + h)) * HD * L;

    // Q tile (16 x 64) as two A operands, loaded once per wave
    v16bf aq0 = load_a_tile(Qh, HD, q0, 0, lane);
    v16bf aq1 = load_a_tile(Qh, HD, q0, 32, lane);

    const float NEG_INF = -__builtin_inff();
    float mrow[8], lrow[8];
#pragma unroll
    for (int i = 0; i < 8; i++) { mrow[i] = NEG_INF; lrow[i] = 0.0f; }
    v8f acc[4] = {};
    const float sc = 0.125f;  // 1/sqrt(64)

    // stage one 32-key block: K block is a contiguous 4KB run in (L,HD) layout;
    // V block is 64 rows x 64B from the (HD,L) layout.  2+2 async ops per wave.
    unsigned kbase0 = (unsigned)(size_t)(void*)&sK[0][0];
    unsigned kbase1 = (unsigned)(size_t)(void*)&sK[1][0];
    unsigned vbase0 = (unsigned)(size_t)(void*)&sV[0][0];
    unsigned vbase1 = (unsigned)(size_t)(void*)&sV[1][0];

    auto stage_kv = [&](int m0, unsigned kb, unsigned vb) {
        const char* gk = (const char*)(Kh + (size_t)m0 * HD);
        async_cp16(kb + (unsigned)tid * 16u, gk + (size_t)tid * 16);
        async_cp16(kb + (unsigned)tid * 16u + 2048u, gk + (size_t)tid * 16 + 2048);
#pragma unroll
        for (int c = 0; c < 2; c++) {
            int ch = tid + c * 128;          // 256 chunks of 16B
            int e = ch >> 2, seg = ch & 3;   // row in (HD,32), 16B segment
            const char* gv = (const char*)Vh + ((size_t)e * L + m0) * 2 + seg * 16;
            async_cp16(vb + (unsigned)ch * 16u, gv);
        }
    };

    constexpr int NBLK = L / 32;   // 64
    stage_kv(0, kbase0, vbase0);

    for (int it = 0; it < NBLK; ++it) {
        int cur = it & 1;
        unsigned kb = cur ? kbase1 : kbase0;
        unsigned vb = cur ? vbase1 : vbase0;
        if (it + 1 < NBLK) {
            stage_kv((it + 1) * 32, cur ? kbase0 : kbase1, cur ? vbase0 : vbase1);
            asm volatile("s_wait_asynccnt 4" ::: "memory");   // current block's 4 ops done
        } else {
            asm volatile("s_wait_asynccnt 0" ::: "memory");
        }
        __syncthreads();   // staged K/V visible to all waves

        const unsigned short* kbuf = cur ? &sK[1][0] : &sK[0][0];
        const unsigned short* vbuf = cur ? &sV[1][0] : &sV[0][0];
        int m0 = it * 32;

        // S = Q K^T for 16 q-rows x 32 keys (two 16x16 C tiles, K-dim 64)
        v8f s0 = {}, s1 = {};
        s0 = wmma_bf16(aq0, load_b_tile(kbuf, HD, 0, 0, lane), s0);
        s0 = wmma_bf16(aq1, load_b_tile(kbuf, HD, 0, 32, lane), s0);
        s1 = wmma_bf16(aq0, load_b_tile(kbuf, HD, 16, 0, lane), s1);
        s1 = wmma_bf16(aq1, load_b_tile(kbuf, HD, 16, 32, lane), s1);

        int mk0 = mask[b * L + m0 + ln];
        int mk1 = mask[b * L + m0 + 16 + ln];

        float alpha[8];
#pragma unroll
        for (int i = 0; i < 8; i++) {
            float x0 = mk0 ? s0[i] * sc : -1e9f;
            float x1 = mk1 ? s1[i] * sc : -1e9f;
            // row max across the 16 lanes holding this row (butterfly stays in half)
            float t = fmaxf(x0, x1);
            t = fmaxf(t, __shfl_xor(t, 1, 32));
            t = fmaxf(t, __shfl_xor(t, 2, 32));
            t = fmaxf(t, __shfl_xor(t, 4, 32));
            t = fmaxf(t, __shfl_xor(t, 8, 32));
            float mn = fmaxf(mrow[i], t);
            float al = (mrow[i] == NEG_INF) ? 0.0f : __expf(mrow[i] - mn);
            float p0 = (mn == NEG_INF) ? 0.0f : __expf(x0 - mn);
            float p1 = (mn == NEG_INF) ? 0.0f : __expf(x1 - mn);
            float rs = p0 + p1;
            rs += __shfl_xor(rs, 1, 32);
            rs += __shfl_xor(rs, 2, 32);
            rs += __shfl_xor(rs, 4, 32);
            rs += __shfl_xor(rs, 8, 32);
            lrow[i] = lrow[i] * al + rs;
            mrow[i] = mn;
            alpha[i] = al;
            int row = i + half * 8;
            sP[wave][row * 32 + ln] = f2bf(p0);
            sP[wave][row * 32 + ln + 16] = f2bf(p1);
        }
#pragma unroll
        for (int j = 0; j < 4; j++)
#pragma unroll
            for (int i = 0; i < 8; i++) acc[j][i] *= alpha[i];

        // P is wave-private; DS ops from one wave are in-order, so the A-layout
        // reload below observes the stores above without a barrier.
        v16bf pA = load_a_tile(&sP[wave][0], 32, 0, 0, lane);
#pragma unroll
        for (int j = 0; j < 4; j++) {
            acc[j] = wmma_bf16(pA, load_b_tile(vbuf, 32, j * 16, 0, lane), acc[j]);
        }
        __syncthreads();   // all waves done with this K/V buffer before re-staging
    }

    float inv[8];
#pragma unroll
    for (int i = 0; i < 8; i++) inv[i] = (lrow[i] > 0.0f) ? 1.0f / lrow[i] : 0.0f;
#pragma unroll
    for (int j = 0; j < 4; j++)
#pragma unroll
        for (int i = 0; i < 8; i++) {
            int q = q0 + i + half * 8;
            concat[((size_t)(b * L + q)) * D + h * HD + j * 16 + ln] = f2bf(acc[j][i] * inv[i]);
        }
}

// ---------------------------------------------------------------------------
// Stage 3: output projection (B*L, D) @ Wo (D, D) + bo  -> f32 out
// ---------------------------------------------------------------------------
__global__ __launch_bounds__(32) void proj_kernel(
    const unsigned short* __restrict__ concat,   // (B*L, D) bf16
    const unsigned short* __restrict__ woT,      // (Dout, Din) bf16
    const float* __restrict__ bo,
    float* __restrict__ out) {
    int lane = threadIdx.x;
    int ln = lane & 15, half = lane >> 4;
    int row0 = blockIdx.x * 16;
    int n0 = blockIdx.y * 64;

    v8f acc[4] = {};
    for (int k0 = 0; k0 < D; k0 += 32) {
        v16bf a = load_a_tile(concat, D, row0, k0, lane);
#pragma unroll
        for (int j = 0; j < 4; j++) {
            acc[j] = wmma_bf16(a, load_b_tile(woT, D, n0 + j * 16, k0, lane), acc[j]);
        }
    }
#pragma unroll
    for (int j = 0; j < 4; j++)
#pragma unroll
        for (int i = 0; i < 8; i++) {
            int col = n0 + j * 16 + ln;
            out[((size_t)(row0 + i + half * 8)) * D + col] = acc[j][i] + bo[col];
        }
}

// ---------------------------------------------------------------------------
extern "C" void kernel_launch(void* const* d_in, const int* in_sizes, int n_in,
                              void* d_out, int out_size, void* d_ws, size_t ws_size,
                              hipStream_t stream) {
    const float* x    = (const float*)d_in[0];
    const int*   msk  = (const int*)d_in[1];
    const float* Wq   = (const float*)d_in[2];
    const float* bq   = (const float*)d_in[3];
    const float* Wk   = (const float*)d_in[4];
    const float* bk   = (const float*)d_in[5];
    const float* Wv   = (const float*)d_in[6];
    const float* bv   = (const float*)d_in[7];
    const float* Wo   = (const float*)d_in[8];
    const float* bo   = (const float*)d_in[9];
    float* out = (float*)d_out;

    char* ws = (char*)d_ws;
    size_t off = 0;
    auto take = [&](size_t bytes) -> void* {
        void* p = ws + off;
        off += (bytes + 255) & ~(size_t)255;
        return p;
    };
    unsigned short* xb   = (unsigned short*)take((size_t)BATCH * L * D * 2);
    unsigned short* wqT  = (unsigned short*)take((size_t)H * HD * D * 2);
    unsigned short* wkT  = (unsigned short*)take((size_t)H * HD * D * 2);
    unsigned short* wvT  = (unsigned short*)take((size_t)H * HD * D * 2);
    unsigned short* woT  = (unsigned short*)take((size_t)D * D * 2);
    unsigned short* Qb   = (unsigned short*)take((size_t)BATCH * H * L * HD * 2);
    unsigned short* Kb   = (unsigned short*)take((size_t)BATCH * H * L * HD * 2);
    unsigned short* Vt   = (unsigned short*)take((size_t)BATCH * H * L * HD * 2);
    unsigned short* cat  = (unsigned short*)take((size_t)BATCH * L * D * 2);

    // Stage 0: conversions
    {
        int n = BATCH * L * D;
        cvt_f32_to_bf16<<<(n + 255) / 256, 256, 0, stream>>>(x, xb, n);
        int nw = H * D * HD;
        transpose_to_bf16<<<(nw + 255) / 256, 256, 0, stream>>>(Wq, wqT, H, D, HD);
        transpose_to_bf16<<<(nw + 255) / 256, 256, 0, stream>>>(Wk, wkT, H, D, HD);
        transpose_to_bf16<<<(nw + 255) / 256, 256, 0, stream>>>(Wv, wvT, H, D, HD);
        int nwo = D * D;
        transpose_to_bf16<<<(nwo + 255) / 256, 256, 0, stream>>>(Wo, woT, 1, D, D);
    }
    // Stage 1: QKV projection
    qkv_kernel<<<dim3(BATCH * L / 16, H), 32, 0, stream>>>(xb, wqT, wkT, wvT, bq, bk, bv,
                                                           Qb, Kb, Vt);
    // Stage 2: flash attention (4 waves/WG, async K/V staging)
    attn_kernel<<<dim3(L / 64, H, BATCH), 128, 0, stream>>>(Qb, Kb, Vt, msk, cat);
    // Stage 3: output projection
    proj_kernel<<<dim3(BATCH * L / 16, D / 64), 32, 0, stream>>>(cat, woT, bo, out);
}
